// Neural_LP_Learner_60816736911569
// MI455X (gfx1250) — compile-verified, matrix operand order checked by reading, not run on hardware
//
#include <hip/hip_runtime.h>
#include <math.h>

// ---------------------------------------------------------------------------
// Problem constants (from the reference)
// ---------------------------------------------------------------------------
#define T_STEPS 4
#define R_DIM   64
#define E_DIM   1024
#define B_DIM   128
#define IN_DIM  2112         // R + 2E
#define G_DIM   256          // 4*R

typedef __attribute__((ext_vector_type(16))) __bf16 v16bf;
typedef __attribute__((ext_vector_type(8)))  float  v8f;

union ABFrag { uint4 u[2]; v16bf v; };

__device__ __forceinline__ unsigned short f2bf(float f) {
    unsigned u = __float_as_uint(f);
    unsigned r = (u + 0x7FFFu + ((u >> 16) & 1u)) >> 16;  // round-to-nearest-even
    return (unsigned short)r;
}
__device__ __forceinline__ float sigm(float x) { return 1.0f / (1.0f + __expf(-x)); }

// ---------------------------------------------------------------------------
// K1: kb fp32 -> bf16 (so the 4 GEMM passes stream 128MB, L2-resident)
// ---------------------------------------------------------------------------
__global__ void __launch_bounds__(256) k_convert_kb(const float* __restrict__ kb,
                                                    unsigned short* __restrict__ out) {
    size_t i = (size_t)blockIdx.x * blockDim.x + threadIdx.x;  // one float4 per thread
    float4 v = ((const float4*)kb)[i];
    ushort4 o;
    o.x = f2bf(v.x); o.y = f2bf(v.y); o.z = f2bf(v.z); o.w = f2bf(v.w);
    ((ushort4*)out)[i] = o;
}

// ---------------------------------------------------------------------------
// K2: full 4-layer LSTM stack + h = softmax(hs) + h2 = softmax(h)
// One 256-thread workgroup per batch element (batch-independent recurrence).
// ---------------------------------------------------------------------------
__global__ void __launch_bounds__(256) k_lstm(
    const float* __restrict__ x,
    const float* __restrict__ Wih0, const float* __restrict__ Whh0,
    const float* __restrict__ bih0, const float* __restrict__ bhh0,
    const float* __restrict__ Wih,  const float* __restrict__ Whh,
    const float* __restrict__ bih,  const float* __restrict__ bhh,
    float* __restrict__ h_out, float* __restrict__ h2_out)
{
    __shared__ float sh[R_DIM], sc[R_DIM], zbuf[G_DIM];
    __shared__ float hsA[T_STEPS][R_DIM], hsB[T_STEPS][R_DIM];

    const int b = blockIdx.x;
    const int g = threadIdx.x;
    const float* xb = x + (size_t)b * IN_DIM;

    // ---- layer 0: pre (same for every t since x_seq is broadcast) ----
    float p = bih0[g] + bhh0[g];
    {
        const float* w = Wih0 + (size_t)g * IN_DIM;
        for (int i = 0; i < IN_DIM; ++i) p += xb[i] * w[i];
    }
    if (g < R_DIM) { sh[g] = 0.f; sc[g] = 0.f; }
    __syncthreads();
    for (int t = 0; t < T_STEPS; ++t) {
        float z = p;
        const float* wh = Whh0 + g * R_DIM;
        for (int j = 0; j < R_DIM; ++j) z += sh[j] * wh[j];
        zbuf[g] = z;
        __syncthreads();
        if (g < R_DIM) {
            float iv = sigm(zbuf[g]);
            float fv = sigm(zbuf[R_DIM + g]);
            float gv = tanhf(zbuf[2 * R_DIM + g]);
            float ov = sigm(zbuf[3 * R_DIM + g]);
            float c  = fv * sc[g] + iv * gv;
            float hh = ov * tanhf(c);
            sc[g] = c; sh[g] = hh; hsA[t][g] = hh;
        }
        __syncthreads();
    }

    // ---- layers 1..3 ----
    float (*cur)[R_DIM] = hsA;
    float (*nxt)[R_DIM] = hsB;
    for (int l = 0; l < T_STEPS - 1; ++l) {
        const float* WihL = Wih + (size_t)l * G_DIM * R_DIM;
        const float* WhhL = Whh + (size_t)l * G_DIM * R_DIM;
        float bb = bih[l * G_DIM + g] + bhh[l * G_DIM + g];
        float p4[T_STEPS];
        for (int t = 0; t < T_STEPS; ++t) {
            float s = bb;
            const float* wl = WihL + g * R_DIM;
            for (int j = 0; j < R_DIM; ++j) s += cur[t][j] * wl[j];
            p4[t] = s;
        }
        __syncthreads();
        if (g < R_DIM) { sh[g] = 0.f; sc[g] = 0.f; }
        __syncthreads();
        for (int t = 0; t < T_STEPS; ++t) {
            float z = p4[t];
            const float* wh = WhhL + g * R_DIM;
            for (int j = 0; j < R_DIM; ++j) z += sh[j] * wh[j];
            zbuf[g] = z;
            __syncthreads();
            if (g < R_DIM) {
                float iv = sigm(zbuf[g]);
                float fv = sigm(zbuf[R_DIM + g]);
                float gv = tanhf(zbuf[2 * R_DIM + g]);
                float ov = sigm(zbuf[3 * R_DIM + g]);
                float c  = fv * sc[g] + iv * gv;
                float hh = ov * tanhf(c);
                sc[g] = c; sh[g] = hh; nxt[t][g] = hh;
            }
            __syncthreads();
        }
        float (*tmp)[R_DIM] = cur; cur = nxt; nxt = tmp;
    }

    // ---- h = softmax(hs, -1); h2 = softmax(h, -1). 4 groups of 64 threads ----
    {
        const int t = g >> 6, r = g & 63;
        const int base = t * R_DIM;
        float m = -1e30f;
        for (int j = 0; j < R_DIM; ++j) m = fmaxf(m, cur[t][j]);
        float ssum = 0.f;
        for (int j = 0; j < R_DIM; ++j) ssum += __expf(cur[t][j] - m);
        float hv = __expf(cur[t][r] - m) / ssum;
        h_out[((size_t)t * B_DIM + b) * R_DIM + r] = hv;
        zbuf[g] = hv;
        __syncthreads();
        float m2 = -1e30f;
        for (int j = 0; j < R_DIM; ++j) m2 = fmaxf(m2, zbuf[base + j]);
        float s2 = 0.f;
        for (int j = 0; j < R_DIM; ++j) s2 += __expf(zbuf[base + j] - m2);
        h2_out[((size_t)t * B_DIM + b) * R_DIM + r] = __expf(hv - m2) / s2;
    }
}

// ---------------------------------------------------------------------------
// K3 (per iteration i): attention over mem stack -> prev, then materialize
// A-matrix prev2[b, r*1024+f] = h2[i,b,r] * prev[b,f]  (bf16, 16 MB)
// ---------------------------------------------------------------------------
__global__ void __launch_bounds__(256) k_prev(
    const float* __restrict__ x, const float* __restrict__ h,
    const float* __restrict__ h2, const float* __restrict__ mem,
    unsigned short* __restrict__ prev2, int iter)
{
    const int b = blockIdx.x, tid = threadIdx.x;
    const float* hi = h + ((size_t)iter * B_DIM + b) * R_DIM;

    float att[T_STEPS];
    float mx = -1e30f;
    for (int k = 0; k <= iter; ++k) {
        const float* hk = h + ((size_t)k * B_DIM + b) * R_DIM;
        float d = 0.f;
        for (int r = 0; r < R_DIM; ++r) d += hk[r] * hi[r];
        att[k] = d; mx = fmaxf(mx, d);
    }
    float ssum = 0.f;
    for (int k = 0; k <= iter; ++k) { att[k] = __expf(att[k] - mx); ssum += att[k]; }
    const float inv = 1.f / ssum;

    const int f0 = tid * 4;                       // 1024/256 = 4 f-values per thread
    float pv[4] = {0.f, 0.f, 0.f, 0.f};
    for (int k = 0; k <= iter; ++k) {
        const float* mk = (k == 0) ? (x + (size_t)b * IN_DIM + R_DIM)
                                   : (mem + ((size_t)(k - 1) * B_DIM + b) * E_DIM);
        float a = att[k] * inv;
        for (int q = 0; q < 4; ++q) pv[q] += a * mk[f0 + q];
    }

    const float* h2i = h2 + ((size_t)iter * B_DIM + b) * R_DIM;
    unsigned short* dst = prev2 + (size_t)b * (R_DIM * E_DIM);
    for (int r = 0; r < R_DIM; ++r) {
        float s = h2i[r];
        ushort4 o;
        o.x = f2bf(s * pv[0]); o.y = f2bf(s * pv[1]);
        o.z = f2bf(s * pv[2]); o.w = f2bf(s * pv[3]);
        *(ushort4*)(dst + (size_t)r * E_DIM + f0) = o;
    }
}

// ---------------------------------------------------------------------------
// K4: the big WMMA GEMM.  C(128x1024) = prev2(128x65536) x kb_bf16^T
// Grid: 32 N-columns x 8 K-chunks = 256 WGs, 8 waves each.
// Wave: one 16-row M tile, two 16-col N tiles; K loop 8 r-values x 32-f steps.
// kb's [r][e][f] layout gives B fragments as contiguous b128 loads per lane.
// Deterministic K-split: each chunk writes its own partial buffer.
// ---------------------------------------------------------------------------
__global__ void __launch_bounds__(256) k_gemm(
    const unsigned short* __restrict__ kbbf,
    const unsigned short* __restrict__ prev2,
    float* __restrict__ partial)
{
    const int wgN  = blockIdx.x & 31;     // N column: 32 cols of 32
    const int kc   = blockIdx.x >> 5;     // K chunk: 0..7 (8 r-values each)
    const int n0   = wgN * 32;
    const int r0   = kc * 8;
    const int wave = threadIdx.x >> 5;
    const int lane = threadIdx.x & 31;
    const int laneLo = lane & 15;
    const int s    = lane >> 4;           // lane-half selector (ISA K-striping)
    const int m0   = wave * 16;

    const unsigned short* Abase = prev2 + (size_t)(m0 + laneLo) * (R_DIM * E_DIM);
    const size_t e0 = (size_t)(n0 + laneLo);

    v8f acc0, acc1;
    for (int q = 0; q < 8; ++q) { acc0[q] = 0.f; acc1[q] = 0.f; }

    for (int r = r0; r < r0 + 8; ++r) {
        const unsigned short* Ar = Abase + (size_t)r * E_DIM;
        const unsigned short* B0 = kbbf + ((size_t)r * E_DIM + e0) * E_DIM;
        const unsigned short* B1 = B0 + (size_t)16 * E_DIM;
        for (int fc = 0; fc < E_DIM; fc += 32) {
            ABFrag a, b0, b1;
            a.u[0]  = *(const uint4*)(Ar + fc + s * 8);
            a.u[1]  = *(const uint4*)(Ar + fc + 16 + s * 8);
            b0.u[0] = *(const uint4*)(B0 + fc + s * 8);
            b0.u[1] = *(const uint4*)(B0 + fc + 16 + s * 8);
            b1.u[0] = *(const uint4*)(B1 + fc + s * 8);
            b1.u[1] = *(const uint4*)(B1 + fc + 16 + s * 8);
            acc0 = __builtin_amdgcn_wmma_f32_16x16x32_bf16(
                       false, a.v, false, b0.v, (short)0, acc0, false, false);
            acc1 = __builtin_amdgcn_wmma_f32_16x16x32_bf16(
                       false, a.v, false, b1.v, (short)0, acc1, false, false);
        }
    }

    // C/D layout: VGPR v, lane l -> (m = m0 + v + s*8, n = laneLo)
    float* P = partial + (size_t)kc * (B_DIM * E_DIM);
    for (int v = 0; v < 8; ++v) {
        P[(size_t)(m0 + v + s * 8) * E_DIM + n0 + laneLo]      = acc0[v];
        P[(size_t)(m0 + v + s * 8) * E_DIM + n0 + 16 + laneLo] = acc1[v];
    }
}

// ---------------------------------------------------------------------------
// K5: fixed-order reduction of the 8 K-split partials -> mem[i+1]
// ---------------------------------------------------------------------------
__global__ void __launch_bounds__(256) k_reduce(const float* __restrict__ partial,
                                               float* __restrict__ out) {
    size_t i = (size_t)blockIdx.x * blockDim.x + threadIdx.x;
    float s = 0.f;
    for (int kc = 0; kc < 8; ++kc) s += partial[(size_t)kc * (B_DIM * E_DIM) + i];
    out[i] = s;
}

// ---------------------------------------------------------------------------
// K6: score[b] = sigmoid(-dot(mem[4][b], tail[b]))
// ---------------------------------------------------------------------------
__global__ void k_score(const float* __restrict__ mem4, const float* __restrict__ x,
                        float* __restrict__ out) {
    const int b = threadIdx.x;
    const float* m = mem4 + (size_t)b * E_DIM;
    const float* t = x + (size_t)b * IN_DIM + R_DIM + E_DIM;
    float d = 0.f;
    for (int e = 0; e < E_DIM; ++e) d += m[e] * t[e];
    out[b] = 1.f / (1.f + __expf(d));   // sigmoid(-d)
}

// ---------------------------------------------------------------------------
// Host-side orchestration (all on `stream`, graph-capture safe)
// ---------------------------------------------------------------------------
extern "C" void kernel_launch(void* const* d_in, const int* in_sizes, int n_in,
                              void* d_out, int out_size, void* d_ws, size_t ws_size,
                              hipStream_t stream)
{
    const float* x    = (const float*)d_in[0];
    const float* kb   = (const float*)d_in[1];
    const float* Wih0 = (const float*)d_in[2];
    const float* Whh0 = (const float*)d_in[3];
    const float* bih0 = (const float*)d_in[4];
    const float* bhh0 = (const float*)d_in[5];
    const float* Wih  = (const float*)d_in[6];
    const float* Whh  = (const float*)d_in[7];
    const float* bih  = (const float*)d_in[8];
    const float* bhh  = (const float*)d_in[9];
    float* out = (float*)d_out;

    // Workspace layout (~151 MB):
    //   kb_bf16 : 64M bf16  = 134217728 B
    //   prev2   : 8M  bf16  =  16777216 B
    //   h, h2   : 32768 f32 each
    //   mem[4]  : 4 * 131072 f32
    //   partial : 8 * 131072 f32
    char* ws = (char*)d_ws;
    unsigned short* kbbf  = (unsigned short*)ws;
    unsigned short* prev2 = (unsigned short*)(ws + 134217728ull);
    float* h_buf   = (float*)(ws + 134217728ull + 16777216ull);
    float* h2_buf  = h_buf  + 32768;
    float* mem_buf = h2_buf + 32768;                 // 4 x (128*1024)
    float* partial = mem_buf + 4 * 131072;           // 8 x (128*1024)

    k_convert_kb<<<65536, 256, 0, stream>>>(kb, kbbf);
    k_lstm<<<B_DIM, 256, 0, stream>>>(x, Wih0, Whh0, bih0, bhh0,
                                      Wih, Whh, bih, bhh, h_buf, h2_buf);
    for (int i = 0; i < T_STEPS; ++i) {
        k_prev<<<B_DIM, 256, 0, stream>>>(x, h_buf, h2_buf, mem_buf, prev2, i);
        k_gemm<<<256, 256, 0, stream>>>(kbbf, prev2, partial);
        k_reduce<<<512, 256, 0, stream>>>(partial, mem_buf + (size_t)i * 131072);
    }
    k_score<<<1, B_DIM, 0, stream>>>(mem_buf + 3ull * 131072, x, out);
}
